// PitchShift_1571958030982
// MI455X (gfx1250) — matrix operand order repaired and checked by reading, not compiled, not run on hardware
//
#include <hip/hip_runtime.h>

#define NTHR 256
#define NWIN_CAP 192
#define SPAN_CAP 576
#define WREF_CAP 384
#define PK_CAP   65536
#define SRATE    22050.0

typedef _Float16 v16h __attribute__((ext_vector_type(16)));
typedef float    v8f  __attribute__((ext_vector_type(8)));

struct Scal { unsigned gp_bits; unsigned minpos_bits; float median; int any; };

struct __align__(32) ShMem {
  _Float16 refh[WREF_CAP];     // offset 0: 32B-aligned for v16h B-fragment loads
  _Float16 sigh[SPAN_CAP];
  float normArr[NWIN_CAP];
  float dotArr[NWIN_CAP];
  float pA[SPAN_CAP + 1];
  float pB[SPAN_CAP + 1];
  float rv[NTHR];
  int   ri[NTHR];
  int   found;
};

struct FPRes { int w; float corr; int inew; float peak; };

// ---------------- block-wide helpers (uniform control flow only) -------------

__device__ float blk_sum(float v, float* s, int tid) {
  s[tid] = v; __syncthreads();
  for (int st = NTHR / 2; st > 0; st >>= 1) {
    if (tid < st) s[tid] += s[tid + st];
    __syncthreads();
  }
  float r = s[0]; __syncthreads(); return r;
}

__device__ float blk_max(float v, float* s, int tid) {
  s[tid] = v; __syncthreads();
  for (int st = NTHR / 2; st > 0; st >>= 1) {
    if (tid < st) s[tid] = fmaxf(s[tid], s[tid + st]);
    __syncthreads();
  }
  float r = s[0]; __syncthreads(); return r;
}

// first-occurrence argmax: ties resolved toward the smaller index
__device__ void blk_argmax(float v, int idx, float* sv, int* si, int tid,
                           float* ov, int* oi) {
  sv[tid] = v; si[tid] = idx; __syncthreads();
  for (int st = NTHR / 2; st > 0; st >>= 1) {
    if (tid < st) {
      float bv = sv[tid + st]; int bi = si[tid + st];
      if (bv > sv[tid] || (bv == sv[tid] && bi < si[tid])) { sv[tid] = bv; si[tid] = bi; }
    }
    __syncthreads();
  }
  *ov = sv[0]; *oi = si[0]; __syncthreads();
}

__device__ float arr_at(const float* a, int i, int T) {
  if (i < 0) i = 0; if (i > T - 1) i = T - 1; return a[i];
}

// first j in [start,T) with (wantpos ? a[j]>0 : a[j]<=0), else -1
__device__ int ff(const float* a, int start, int T, bool wantpos, int tid, ShMem* sh) {
  for (int base = start; base < T; base += 1024) {
    if (tid == 0) sh->found = 0x7fffffff;
    __syncthreads();
    int lim = base + 1024; if (lim > T) lim = T;
    int loc = 0x7fffffff;
    for (int j = base + tid; j < lim; j += NTHR) {
      float v = a[j];
      bool hit = wantpos ? (v > 0.f) : !(v > 0.f);
      if (hit) { loc = j; break; }
    }
    if (loc != 0x7fffffff) atomicMin(&sh->found, loc);
    __syncthreads();
    int f = sh->found;
    __syncthreads();
    if (f != 0x7fffffff) return f;
  }
  return -1;
}

__device__ bool find_voiced(const float* a, int i, int T, int tid, ShMem* sh,
                            int* L, int* R) {
  int first = ff(a, i + 1, T, true, tid, sh);
  if (first < 0) return false;
  int e = ff(a, first, T, false, tid, sh);
  if (e < 0) e = T;
  *L = first; *R = e; return true;
}

// ---------------- find_peak: WMMA-accelerated normalized correlation --------

__device__ FPRes find_peak(const float* __restrict__ snd, const float* __restrict__ f0,
                           int T, int i, bool right_dir, int tid, ShMem* sh) {
  FPRes res;
  double fv = (double)arr_at(f0, i, T);
  double fc = fv > 60.0 ? fv : 60.0;
  int w = (int)(SRATE / fc);                // <= 367
  res.w = w; res.corr = -1.0f; res.inew = i; res.peak = 0.f;
  int s = i - w / 2; if (s < 0) s = 0;
  int cand_l, cand_r;
  if (right_dir) {
    cand_l = (int)((double)i + 0.3  * (double)w);
    cand_r = (int)((double)i + 0.75 * (double)w);
  } else {
    cand_l = (int)((double)i - 1.75 * (double)w); if (cand_l < 0) cand_l = 0;
    cand_r = (int)((double)i - 1.3  * (double)w); if (cand_r < 0) cand_r = 0;
  }
  if (cand_l == cand_r || (T - cand_l) < w) return res;
  int len = cand_r + w; if (len > T) len = T; len -= cand_l;
  int nwin = len - w + 1;
  if (nwin < 1) return res;
  if (nwin > NWIN_CAP) nwin = NWIN_CAP;
  if (nwin > SPAN_CAP - w + 1) nwin = SPAN_CAP - w + 1;
  int span = nwin - 1 + w;          // cand_l + span <= T by construction

  // stage f16 signal span (unguarded: in-bounds by construction) + zero tail
  for (int j = tid; j < span; j += NTHR) sh->sigh[j] = (_Float16)snd[cand_l + j];
  for (int j = span + tid; j < SPAN_CAP; j += NTHR) sh->sigh[j] = (_Float16)0.f;
  // stage reference, zero-padded to full capacity (B side must be exact zeros)
  for (int k = tid; k < WREF_CAP; k += NTHR) {
    int g = s + k;
    sh->refh[k] = (_Float16)((k < w && g < T) ? snd[g] : 0.f);
  }
  // squares for prefix-scan window norms
  if (tid == 0) sh->pA[0] = 0.f;
  for (int j = tid; j < span; j += NTHR) {
    float v = snd[cand_l + j];
    sh->pA[j + 1] = v * v;
  }
  // reference norm (f32)
  float racc = 0.f;
  for (int k = tid; k < w; k += NTHR) {
    int g = s + k;
    float v = (g < T) ? snd[g] : 0.f;
    racc += v * v;
  }
  float refnorm = sqrtf(blk_sum(racc, sh->rv, tid));   // internal syncs publish LDS fills

  // Hillis-Steele inclusive scan over pA[0..span]
  float* pin = sh->pA; float* pout = sh->pB;
  int n = span + 1;
  for (int off = 1; off < n; off <<= 1) {
    for (int j = tid; j < n; j += NTHR)
      pout[j] = pin[j] + ((j >= off) ? pin[j - off] : 0.f);
    __syncthreads();
    float* t = pin; pin = pout; pout = t;
  }
  for (int r = tid; r < nwin; r += NTHR) {
    float d = pin[r + w] - pin[r];
    sh->normArr[r] = sqrtf(d > 0.f ? d : 0.f);
  }
  __syncthreads();

  // dots via v_wmma_f32_16x16x32_f16: each wave owns groups of 16 lags,
  // B holds ref replicated across all 16 columns, K accumulated in chunks of 32.
  // All fragment loads are unconditional (zero-padding guarantees correctness).
  int lane = tid & 31;
  int wvid = tid >> 5;
  bool hi  = lane >= 16;
  int mrow = hi ? lane - 16 : lane;
  int nch  = (w + 31) >> 5;
  for (int g = wvid; g * 16 < nwin; g += 8) {
    int m0 = g * 16;
    v8f c = {0.f, 0.f, 0.f, 0.f, 0.f, 0.f, 0.f, 0.f};
    for (int ch = 0; ch < nch; ++ch) {
      int k0 = ch << 5;
      // B fragment: 16 contiguous halves, 32B-aligned -> vector LDS load
      v16h bvv = *(const v16h*)&sh->refh[k0 + (hi ? 16 : 0)];
      // A fragment: sliding rows (2B alignment only) -> plain u16 loads, no guards
      int abase = m0 + mrow + k0 + (hi ? 16 : 0);   // max index 574 < SPAN_CAP
      v16h av;
      #pragma unroll
      for (int e = 0; e < 8; ++e) av[e] = sh->sigh[abase + e];
      #pragma unroll
      for (int e = 0; e < 8; ++e) av[8 + e] = sh->sigh[abase + 8 + e];
      c = __builtin_amdgcn_wmma_f32_16x16x32_f16(false, av, false, bvv,
                                                 (short)0, c, false, false);
    }
    if (lane == 0) {
      #pragma unroll
      for (int r = 0; r < 8; ++r) { int m = m0 + r;     if (m < nwin) sh->dotArr[m] = c[r]; }
    }
    if (lane == 16) {
      #pragma unroll
      for (int r = 0; r < 8; ++r) { int m = m0 + 8 + r; if (m < nwin) sh->dotArr[m] = c[r]; }
    }
  }
  __syncthreads();

  // argmax of normalized correlation (first occurrence)
  float rn = refnorm; if (rn < 1e-12f) rn = 1e-12f;
  float bvv2 = -__builtin_huge_valf(); int bii = 0x7fffffff;
  for (int r = tid; r < nwin; r += NTHR) {
    float nv = sh->normArr[r]; if (nv < 1e-12f) nv = 1e-12f;
    float cv = sh->dotArr[r] / (nv * rn);
    if (cv > bvv2) { bvv2 = cv; bii = r; }
  }
  float cbest; int rbest;
  blk_argmax(bvv2, bii, sh->rv, sh->ri, tid, &cbest, &rbest);

  float pm = 0.f; int b0 = cand_l + rbest;
  for (int k = tid; k < w; k += NTHR) pm = fmaxf(pm, fabsf(snd[b0 + k]));
  float pk = blk_max(pm, sh->rv, tid);

  res.corr = cbest;
  res.inew = i + (rbest + cand_l) - s;
  res.peak = pk;
  return res;
}

// ---------------- misc device helpers ---------------------------------------

__device__ void copy_all(float* out, const float* snd, int T, int tid) {
  for (int k = tid; k < T; k += NTHR) out[k] = snd[k];
}

__device__ void apply_op(float* out, const float* snd, int dst0, int src0,
                         int L, int n, int tid) {
  __syncthreads();
  for (int k = tid; k < L; k += NTHR) {
    float wv = 0.5f - 0.5f * cosf((float)(6.283185307179586 * (double)k / (double)n));
    out[dst0 + k] += wv * snd[src0 + k];
  }
  __threadfence_block();
  __syncthreads();
}

__device__ int nearest_peak(const int* pk, int P, int lv) {
  int lo = 0, hi = P;
  while (lo < hi) { int mid = (lo + hi) >> 1; if (pk[mid] < lv) lo = mid + 1; else hi = mid; }
  int p;
  if (lo == 0) p = 0;
  else if (lo == P) p = P - 1;
  else {
    long long dl = (long long)lv - pk[lo - 1];
    long long dr = (long long)pk[lo] - lv;
    p = (dl <= dr) ? lo - 1 : lo;
  }
  while (p > 0 && pk[p - 1] == pk[p]) --p;   // np.argmin first-occurrence on dups
  return p;
}

// ---------------- kernels ----------------------------------------------------

__global__ void k_init(float* out, int T, Scal* sc) {
  int i = blockIdx.x * blockDim.x + threadIdx.x;
  if (i < T) out[i] = 0.f;
  if (i == 0) { sc->gp_bits = 0u; sc->minpos_bits = 0x7f800000u; sc->median = 1.f; sc->any = 0; }
}

__global__ void __launch_bounds__(NTHR) k_median(const float* pitch, int S, Scal* sc) {
  __shared__ int s_any, s_npos;
  int tid = threadIdx.x;
  if (tid == 0) { s_any = 0; s_npos = 0; }
  __syncthreads();
  int la = 0, lp = 0;
  for (int j = tid; j < S; j += NTHR) {
    float v = pitch[j];
    if (v > 0.f)    lp++;
    if (v > 1e-5f)  la++;
  }
  atomicAdd(&s_npos, lp); atomicAdd(&s_any, la);
  __syncthreads();
  int npos = s_npos;
  if (tid == 0) sc->any = (s_any > 0) ? 1 : 0;
  if (npos > 0) {
    int R = (npos - 1) / 2;
    for (int j = tid; j < S; j += NTHR) {
      float v = pitch[j];
      if (!(v > 0.f)) continue;
      int less = 0, eq = 0;
      for (int q = 0; q < S; ++q) {
        float u = pitch[q];
        if (u > 0.f) { if (u < v) less++; else if (u == v) eq++; }
      }
      if (R >= less && R < less + eq) sc->median = v;  // all writers agree
    }
  }
}

__global__ void k_interp(const float* __restrict__ pitch, int S, int T,
                         const int* stepsPtr, const float* prPtr, const Scal* sc,
                         float* __restrict__ f0, float* __restrict__ f0s) {
  long i = (long)blockIdx.x * blockDim.x + threadIdx.x;
  if (i >= T) return;
  double x = ((double)i + 0.5) * ((double)S / (double)T) - 0.5;
  if (x < 0.0) x = 0.0;
  double smax = (double)(S - 1);
  if (x > smax) x = smax;
  int i0 = (int)x;
  int i1 = i0 + 1; if (i1 > S - 1) i1 = S - 1;
  float fr = (float)(x - (double)i0);
  float v = pitch[i0] * (1.0f - fr) + pitch[i1] * fr;
  f0[i] = v;
  double factor = exp2((double)(*stepsPtr) / 12.0);
  double med = (double)sc->median * factor;
  double pr  = (double)(*prPtr);
  f0s[i] = (v > 0.f) ? (float)(med + ((double)v * factor - med) * pr) : 0.f;
}

__global__ void k_reduce(const float* __restrict__ snd, const float* __restrict__ f0s,
                         int T, Scal* sc) {
  int i = blockIdx.x * blockDim.x + threadIdx.x;
  if (i >= T) return;
  atomicMax(&sc->gp_bits, __float_as_uint(fabsf(snd[i])));
  float v = f0s[i];
  if (v > 0.f) atomicMin(&sc->minpos_bits, __float_as_uint(v));
}

__global__ void __launch_bounds__(NTHR)
k_driver(const float* __restrict__ snd, const float* __restrict__ f0,
         const float* __restrict__ f0s, int T,
         int* __restrict__ peaks, int* __restrict__ peaks2,
         Scal* sc, float* __restrict__ out) {
  __shared__ ShMem sh;
  int tid = threadIdx.x;

  if (!sc->any) { copy_all(out, snd, T, tid); return; }
  float gp = __uint_as_float(sc->gp_bits);

  // ---------------- _find_allpeaks (original f0) ----------------
  double added_right = -1e308;
  int i = 0, P = 0;
  for (int outer = 0; outer < 100000; ++outer) {
    int left, right;
    if (!find_voiced(f0, i, T, tid, &sh, &left, &right)) break;
    int middle = (left + right) / 2;
    double fm = (double)arr_at(f0, middle, T);
    double wd = SRATE / fm;
    int wM = (wd > 2.0e9) ? 2000000000 : (int)wd;
    int s0 = middle - wM / 2; if (s0 < 0) s0 = 0;
    int nseg = T - s0; if (wM < nseg) nseg = wM;

    float lmn = __builtin_huge_valf(), lmx = -__builtin_huge_valf();
    int lin = 0x7fffffff, lix = 0x7fffffff;
    for (int k = tid; k < nseg; k += NTHR) {
      float v = snd[s0 + k];
      if (v < lmn) { lmn = v; lin = k; }
      if (v > lmx) { lmx = v; lix = k; }
    }
    float mnv, mxv; int mni, mxi;
    blk_argmax(-lmn, lin, sh.rv, sh.ri, tid, &mnv, &mni); mnv = -mnv;
    blk_argmax(lmx, lix, sh.rv, sh.ri, tid, &mxv, &mxi);
    if (mnv == mxv) i = middle;
    else i = s0 + ((fabsf(mnv) > fabsf(mxv)) ? mni : mxi);
    int backup = i;

    // left scan
    for (int it = 0; it < 100000; ++it) {
      FPRes fp = find_peak(snd, f0, T, i, false, tid, &sh);
      i = fp.inew;
      if (fp.corr == -1.0f) i -= fp.w;
      if (i < left) {
        if ((double)fp.corr > 0.7 && (double)fp.peak > 0.023333 * (double)gp &&
            ((double)i - added_right > 0.8 * (double)fp.w)) {
          if (P < PK_CAP) { if (tid == 0) peaks[P] = i; P++; }
        }
        break;
      }
      if ((double)fp.corr > 0.3 && (fp.peak == 0.0f || (double)fp.peak > 0.01 * (double)gp)) {
        if ((double)i - added_right > 0.8 * (double)fp.w) {
          if (P < PK_CAP) { if (tid == 0) peaks[P] = i; P++; }
        }
      }
    }
    i = backup;
    // right scan
    for (int it = 0; it < 100000; ++it) {
      FPRes fp = find_peak(snd, f0, T, i, true, tid, &sh);
      i = fp.inew;
      if (fp.corr == -1.0f) i += fp.w;
      if (i >= right) {
        if ((double)fp.corr > 0.7 && (double)fp.peak > 0.023333 * (double)gp) {
          if (P < PK_CAP) { if (tid == 0) peaks[P] = i; P++; }
          added_right = (double)i;
        }
        break;
      }
      if ((double)fp.corr > 0.3 && (fp.peak == 0.0f || (double)fp.peak > 0.01 * (double)gp)) {
        if (P < PK_CAP) { if (tid == 0) peaks[P] = i; P++; }
        added_right = (double)i;
      }
    }
    i = right;
  }

  if (P == 0) { copy_all(out, snd, T, tid); return; }

  // rank sort + clip to [0, T-1]
  __threadfence(); __syncthreads();
  for (int j = tid; j < P; j += NTHR) {
    int v = peaks[j]; int rank = 0;
    for (int q = 0; q < P; ++q) {
      int u = peaks[q];
      rank += (u < v) || (u == v && q < j);
    }
    int cv = v; if (cv < 0) cv = 0; if (cv > T - 1) cv = T - 1;
    peaks2[rank] = cv;
  }
  __threadfence(); __syncthreads();

  // ---------------- _psola_plan + apply (shifted f0s) ----------------
  double minpos = (double)__uint_as_float(sc->minpos_bits);
  double maxw = 1.25 * SRATE / minpos;
  int ii = 0;
  for (int outer = 0; outer < 100000; ++outer) {
    if (ii >= T) break;
    int lv, rv;
    if (!find_voiced(f0s, ii, T, tid, &sh, &lv, &rv)) break;
    if (ii < lv) apply_op(out, snd, ii, ii, lv - ii, lv - ii, tid);
    for (int it = 0; it < 100000 && lv < rv; ++it) {
      int p = nearest_peak(peaks2, P, lv);
      double pv = (double)arr_at(f0s, lv, T); if (pv < 60.0) pv = 60.0;
      int period = (int)(SRATE / pv);
      int lw = period / 2, rw = period / 2;
      if (p > 0) {
        int d = peaks2[p] - peaks2[p - 1];
        if ((double)d <= maxw && d < lw) lw = d;
      }
      if (p < P - 1) {
        int d = peaks2[p + 1] - peaks2[p];
        if ((double)d <= maxw && d < rw) rw = d;
      }
      int li = peaks2[p] - lw; if (li < 0) li = 0;
      int ri = peaks2[p] + rw;
      int ival = (ri - li) / 2;
      if (ival <= 0) break;
      int L;
      if (lv - ival < 0) L = 0;  // python negative-start slice -> empty
      else {
        int e1 = lv + ival; if (e1 > T) e1 = T;
        int len1 = e1 - (lv - ival);
        int len2 = T - li; if (len2 > 2 * ival) len2 = 2 * ival;
        L = len1 < len2 ? len1 : len2;
        if (L < 0) L = 0;
      }
      if (L > 0) apply_op(out, snd, lv - ival, li, L, 2 * ival, tid);
      lv += 2 * ival;
    }
    ii = rv;
  }
  // tail: out[tail:] = snd[tail:]
  for (int k = ii + tid; k < T; k += NTHR) out[k] = snd[k];
}

// ---------------- host launcher ---------------------------------------------

extern "C" void kernel_launch(void* const* d_in, const int* in_sizes, int n_in,
                              void* d_out, int out_size, void* d_ws, size_t ws_size,
                              hipStream_t stream) {
  const float* snd   = (const float*)d_in[0];
  const float* pitch = (const float*)d_in[1];
  const int*   steps = (const int*)d_in[2];
  const float* pr    = (const float*)d_in[3];
  int T = in_sizes[0];
  int S = in_sizes[1];
  float* out = (float*)d_out;

  float* f0     = (float*)d_ws;
  float* f0s    = f0 + T;
  int*   peaks  = (int*)(f0s + T);
  int*   peaks2 = peaks + PK_CAP;
  Scal*  sc     = (Scal*)(peaks2 + PK_CAP);

  int nb = (T + NTHR - 1) / NTHR;
  k_init  <<<nb, NTHR, 0, stream>>>(out, T, sc);
  k_median<<<1,  NTHR, 0, stream>>>(pitch, S, sc);
  k_interp<<<nb, NTHR, 0, stream>>>(pitch, S, T, steps, pr, sc, f0, f0s);
  k_reduce<<<nb, NTHR, 0, stream>>>(snd, f0s, T, sc);
  k_driver<<<1,  NTHR, 0, stream>>>(snd, f0, f0s, T, peaks, peaks2, sc, out);
}
